// WeightedAttention_40346922779022
// MI455X (gfx1250) — matrix operand order repaired and checked by reading, not compiled, not run on hardware
//
#include <hip/hip_runtime.h>
#include <hip/hip_bf16.h>

// Problem constants
#define BATCH 2048
#define SEQ   65
#define SW    64      // walk length (S-1)
#define IND   256     // IN_DIM (= K of projections)
#define HD    8       // heads
#define LD    256     // L per head
#define HL    2048    // H*L

typedef __attribute__((ext_vector_type(16))) __bf16          bf16x16;
typedef __attribute__((ext_vector_type(8)))  float           floatx8;
typedef __attribute__((ext_vector_type(8)))  unsigned short  ushortx8;
typedef __attribute__((ext_vector_type(16))) unsigned short  ushortx16;

__device__ __forceinline__ unsigned short f2bf(float f) {
  unsigned int u = __float_as_uint(f);
  u += 0x7FFFu + ((u >> 16) & 1u);          // round-to-nearest-even
  return (unsigned short)(u >> 16);
}
__device__ __forceinline__ float bf2f(unsigned short s) {
  return __uint_as_float(((unsigned int)s) << 16);
}
__device__ __forceinline__ float sigmoidf_fast(float x) {
  return 1.0f / (1.0f + __expf(-x));
}

// A-fragment (16x32 bf16): lane = M row (lane&15); lanes 0-15 hold K=0..15,
// lanes 16-31 hold K=16..31 -> one contiguous 32B load at kbase.
__device__ __forceinline__ bf16x16 load_a_frag(const unsigned short* p) {
  return __builtin_bit_cast(bf16x16, *(const ushortx16*)p);
}
// B-fragment (32x16 bf16, column-major storage): lane = N col (lane&15);
// lanes 0-15 hold K=0..15, lanes 16-31 hold K=16..31 -> one contiguous 32B load.
__device__ __forceinline__ bf16x16 load_b_frag(const unsigned short* p) {
  return __builtin_bit_cast(bf16x16, *(const ushortx16*)p);
}

// ---------------- prep: f32 weights -> bf16, K-contiguous layouts ----------------
__global__ void prep_qkv(const float* __restrict__ Wq, const float* __restrict__ Wk,
                         const float* __restrict__ Wv, unsigned short* __restrict__ WqT,
                         unsigned short* __restrict__ WkT, unsigned short* __restrict__ WvT) {
  int hl = blockIdx.x;          // h*256 + l
  int d  = threadIdx.x;         // 0..255
  int h = hl >> 8, l = hl & 255;
  size_t src = ((size_t)h * IND + d) * LD + l;   // W[h][d][l]
  size_t dst = (size_t)hl * IND + d;             // WT[h][l][d]
  WqT[dst] = f2bf(Wq[src]);
  WkT[dst] = f2bf(Wk[src]);
  WvT[dst] = f2bf(Wv[src]);
}

__global__ void prep_O(const float* __restrict__ O, unsigned short* __restrict__ OT) {
  int n = blockIdx.x;                            // 0..255
  for (int k = threadIdx.x; k < HL; k += blockDim.x)
    OT[(size_t)n * HL + k] = f2bf(O[(size_t)k * IND + n]);   // OT[n][k]
}

// ---------------- kernel 1: per-(b,h) fused q/k/v GEMMs + softmax + res ----------
__global__ void __launch_bounds__(256)
attn_kernel(const float* __restrict__ x, const unsigned short* __restrict__ WqT,
            const unsigned short* __restrict__ WkT, const unsigned short* __restrict__ WvT,
            const float* __restrict__ Wsc, const float* __restrict__ P,
            const float* __restrict__ bias, unsigned short* __restrict__ resB) {
  extern __shared__ unsigned char smem[];
  unsigned short* sWalk = (unsigned short*)smem;       // 64x256 bf16
  unsigned short* sOrig = sWalk + SW * IND;            // 64x256 bf16
  unsigned short* sV    = sOrig + SW * IND;            // 64x256 bf16 (sigmoid(v))
  float* sPart = (float*)(sV + SW * LD);               // 8 waves x 64 rows
  float* sProb = sPart + 8 * SW;                       // 64 softmax probs

  const int b = blockIdx.x;
  const int h = blockIdx.y;
  const int tid  = threadIdx.x;
  const int lane = tid & 31;
  const int wv   = tid >> 5;       // wave id 0..7
  const int hf   = lane >> 4;      // half-wave (K-half selector)
  const int ln   = lane & 15;      // M row / N col within tile

  // Stage walk and origin+P rows into LDS as bf16
  const float* xb = x + (size_t)b * SEQ * IND;
  for (int i = tid; i < SW * IND; i += 256) {
    int s = i >> 8, d = i & 255;
    sWalk[i] = f2bf(xb[(s + 1) * IND + d]);
    sOrig[i] = f2bf(xb[d] + P[i]);           // origin[b][d] + P[s][d]
  }
  for (int i = tid; i < 8 * SW; i += 256) sPart[i] = 0.0f;
  __syncthreads();

  const unsigned short* Wqh = WqT + (size_t)h * LD * IND;
  const unsigned short* Wkh = WkT + (size_t)h * LD * IND;
  const unsigned short* Wvh = WvT + (size_t)h * LD * IND;
  const float bh = bias[h];

  float pscore[32];                // per-lane partial score: [m(4)][r(8)]
#pragma unroll
  for (int j = 0; j < 32; ++j) pscore[j] = 0.0f;

  for (int t = 0; t < 2; ++t) {    // each wave owns 2 N-tiles, all 4 M-tiles
    const int nTile = wv * 2 + t;
    const int n = nTile * 16 + ln;
    floatx8 accQ[4], accK[4], accV[4];
#pragma unroll
    for (int m = 0; m < 4; ++m) { accQ[m] = (floatx8)0; accK[m] = (floatx8)0; accV[m] = (floatx8)0; }

    for (int kk = 0; kk < 8; ++kk) {           // K = 256 in steps of 32
      const int kbB = kk * 32 + 16 * hf;
      const int kbA = kk * 32 + 16 * hf;
      bf16x16 bq  = load_b_frag(Wqh + (size_t)n * IND + kbB);
      bf16x16 bk  = load_b_frag(Wkh + (size_t)n * IND + kbB);
      bf16x16 bvv = load_b_frag(Wvh + (size_t)n * IND + kbB);
#pragma unroll
      for (int m = 0; m < 4; ++m) {
        const int row = m * 16 + ln;
        bf16x16 aW = load_a_frag(sWalk + row * IND + kbA);
        bf16x16 aO = load_a_frag(sOrig + row * IND + kbA);
        accQ[m] = __builtin_amdgcn_wmma_f32_16x16x32_bf16(false, aW, false, bq,  (short)0, accQ[m], false, false);
        accK[m] = __builtin_amdgcn_wmma_f32_16x16x32_bf16(false, aO, false, bk,  (short)0, accK[m], false, false);
        accV[m] = __builtin_amdgcn_wmma_f32_16x16x32_bf16(false, aW, false, bvv, (short)0, accV[m], false, false);
      }
    }
    // Fused epilogue: sigmoid, edge = cos(k*q)+bias, score partials, store v
    const float Wl = Wsc[(size_t)h * LD + n];
#pragma unroll
    for (int m = 0; m < 4; ++m) {
#pragma unroll
      for (int r = 0; r < 8; ++r) {
        float qv = sigmoidf_fast(accQ[m][r]);
        float kv = sigmoidf_fast(accK[m][r]);
        float vv = sigmoidf_fast(accV[m][r]);
        float edge = __cosf(kv * qv) + bh;
        pscore[m * 8 + r] += edge * Wl;
        int srow = m * 16 + r + 8 * hf;        // C layout: M = r + 8*half
        sV[srow * LD + n] = f2bf(vv);
      }
    }
  }

  // Deterministic cross-lane reduction of score partials (columns live on ln=0..15)
#pragma unroll
  for (int j = 0; j < 32; ++j) {
    float v = pscore[j];
    v += __shfl_xor(v, 1);
    v += __shfl_xor(v, 2);
    v += __shfl_xor(v, 4);
    v += __shfl_xor(v, 8);                     // stays within each 16-lane half
    pscore[j] = v;
  }
  if (ln == 0) {                               // lanes 0 and 16 write their halves
#pragma unroll
    for (int m = 0; m < 4; ++m)
#pragma unroll
      for (int r = 0; r < 8; ++r)
        sPart[wv * SW + m * 16 + r + 8 * hf] = pscore[m * 8 + r];
  }
  __syncthreads();

  // Softmax over s=64 on wave 0 (wave32)
  if (tid < 32) {
    float s0 = 0.0f, s1 = 0.0f;
    for (int w = 0; w < 8; ++w) { s0 += sPart[w * SW + tid]; s1 += sPart[w * SW + tid + 32]; }
    float mx = fmaxf(s0, s1);
#pragma unroll
    for (int off = 16; off >= 1; off >>= 1) mx = fmaxf(mx, __shfl_xor(mx, off));
    float e0 = __expf(s0 - mx), e1 = __expf(s1 - mx);
    float sum = e0 + e1;
#pragma unroll
    for (int off = 16; off >= 1; off >>= 1) sum += __shfl_xor(sum, off);
    float inv = 1.0f / sum;
    sProb[tid]      = e0 * inv;
    sProb[tid + 32] = e1 * inv;
  }
  __syncthreads();

  // res[l] = sum_s prob[s] * v[s,l]; one column per thread
  {
    float acc = 0.0f;
    const int l = tid;
    for (int s = 0; s < SW; ++s) acc += sProb[s] * bf2f(sV[s * LD + l]);
    resB[(size_t)b * HL + (size_t)h * LD + l] = f2bf(acc);
  }
}

// ---------------- kernel 2: out = sigmoid(res @ O), one 16x16 tile per wave -----
__global__ void __launch_bounds__(256)
out_gemm(const unsigned short* __restrict__ resB, const unsigned short* __restrict__ OT,
         float* __restrict__ out) {
  const int tid  = threadIdx.x;
  const int lane = tid & 31;
  const int wv   = tid >> 5;
  const int hf   = lane >> 4;
  const int ln   = lane & 15;
  const int gid  = blockIdx.x * 8 + wv;        // 256 blocks * 8 waves = 2048 tiles
  const int mTile = gid >> 4;                  // 128 M tiles (B=2048)
  const int nTile = gid & 15;                  // 16 N tiles  (256 cols)
  const int row = mTile * 16 + ln;
  const int n   = nTile * 16 + ln;
  floatx8 acc = (floatx8)0;
  for (int kk = 0; kk < 64; ++kk) {            // K = 2048
    bf16x16 a  = load_a_frag(resB + (size_t)row * HL + kk * 32 + 16 * hf);
    bf16x16 bf = load_b_frag(OT   + (size_t)n   * HL + kk * 32 + 16 * hf);
    acc = __builtin_amdgcn_wmma_f32_16x16x32_bf16(false, a, false, bf, (short)0, acc, false, false);
  }
#pragma unroll
  for (int r = 0; r < 8; ++r) {
    int gm = mTile * 16 + r + 8 * hf;
    out[(size_t)gm * 512 + nTile * 16 + ln] = sigmoidf_fast(acc[r]);
  }
}

// ---------------- kernel 3: out[:, 256:512] = origin ----------------------------
__global__ void copy_origin(const float* __restrict__ x, float* __restrict__ out) {
  int idx = blockIdx.x * 256 + threadIdx.x;
  int b = idx >> 8, d = idx & 255;
  out[(size_t)b * 512 + 256 + d] = x[(size_t)b * SEQ * IND + d];
}

extern "C" void kernel_launch(void* const* d_in, const int* in_sizes, int n_in,
                              void* d_out, int out_size, void* d_ws, size_t ws_size,
                              hipStream_t stream) {
  const float* x    = (const float*)d_in[0];
  const float* Wq   = (const float*)d_in[1];
  const float* Wk   = (const float*)d_in[2];
  const float* Wv   = (const float*)d_in[3];
  const float* Wsc  = (const float*)d_in[4];
  const float* O    = (const float*)d_in[5];
  const float* P    = (const float*)d_in[6];
  const float* bias = (const float*)d_in[7];
  float* out = (float*)d_out;

  // Workspace layout (bf16/ushort elements): ~12.6 MB total
  unsigned short* ws   = (unsigned short*)d_ws;
  unsigned short* WqT  = ws;                              // 8*256*256
  unsigned short* WkT  = WqT + (size_t)HD * LD * IND;
  unsigned short* WvT  = WkT + (size_t)HD * LD * IND;
  unsigned short* OT   = WvT + (size_t)HD * LD * IND;     // 256*2048
  unsigned short* resB = OT  + (size_t)IND * HL;          // 2048*2048

  prep_qkv<<<dim3(HD * LD), dim3(256), 0, stream>>>(Wq, Wk, Wv, WqT, WkT, WvT);
  prep_O<<<dim3(IND), dim3(256), 0, stream>>>(O, OT);

  size_t smem = (size_t)(3 * SW * IND) * sizeof(unsigned short)   // walk/orig/v
              + (size_t)(8 * SW + SW) * sizeof(float);            // partials + probs
  attn_kernel<<<dim3(BATCH, HD), dim3(256), smem, stream>>>(x, WqT, WkT, WvT, Wsc, P, bias, resB);

  out_gemm<<<dim3(256), dim3(256), 0, stream>>>(resB, OT, out);
  copy_origin<<<dim3(BATCH), dim3(256), 0, stream>>>(x, out);
}
